// Controller_40467181863500
// MI455X (gfx1250) — matrix operand order repaired and checked by reading, not compiled, not run on hardware
//
#include <hip/hip_runtime.h>
#include <hip/hip_bf16.h>

#define H 2048
#define H4 8192
#define NUM_LAYERS 24
#define NCELLS 6
#define KSPLIT 4            // waves cooperating on one 16-row tile (K split)
#define KCHUNK (H / KSPLIT) // 512 K per wave

typedef __bf16 bf16_t;
typedef __attribute__((ext_vector_type(16))) __bf16 v16bf;
typedef __attribute__((ext_vector_type(8)))  __bf16 v8bf;
typedef __attribute__((ext_vector_type(8)))  float  v8f;

// ---------------------------------------------------------------------------
// fp32 -> bf16 weight conversion (once per launch; shrinks the L2-resident
// weight set from 161MB fp32 to ~84MB bf16 -> fits easily in 192MB L2 and
// halves the per-step weight stream that bounds each sequential LSTM step)
// ---------------------------------------------------------------------------
__global__ __launch_bounds__(256) void k_convert(const float* __restrict__ src,
                                                 bf16_t* __restrict__ dst, int n) {
  int i = blockIdx.x * blockDim.x + threadIdx.x;
  int stride = gridDim.x * blockDim.x;
  for (; i < n; i += stride) dst[i] = (bf16_t)src[i];
}

// h = c = 0, embed = emb0, bias = b_ih + b_hh, rng seed
__global__ __launch_bounds__(256) void k_init(const float* __restrict__ emb0,
                                              const float* __restrict__ bih,
                                              const float* __restrict__ bhh,
                                              float* __restrict__ h, float* __restrict__ cst,
                                              float* __restrict__ embed,
                                              bf16_t* __restrict__ embed_bf,
                                              bf16_t* __restrict__ hbf,
                                              float* __restrict__ bias,
                                              unsigned* __restrict__ rng) {
  int i = blockIdx.x * blockDim.x + threadIdx.x;
  if (i < H) {
    h[i] = 0.f; cst[i] = 0.f;
    float e = emb0[i];
    embed[i] = e; embed_bf[i] = (bf16_t)e; hbf[i] = (bf16_t)0.f;
  }
  if (i < H4) bias[i] = bih[i] + bhh[i];
  if (i == 0) *rng = 0x2a2a2a2au;
}

// ---------------------------------------------------------------------------
// WMMA GEMV core: one 16-row tile is computed by KSPLIT waves, each owning a
// disjoint K chunk, accumulating with v_wmma_f32_16x16x32_bf16; partials are
// reduced through LDS. B is the vector broadcast into all 16 columns so every
// D column equals W*x; lanes 0-7 / 16-23 hold rows 0-7 / 8-15 of D.
// ---------------------------------------------------------------------------
__device__ __forceinline__ v8f wmma_gemv_chunk(const bf16_t* __restrict__ Wrow,
                                               const bf16_t* __restrict__ xb,
                                               int kBegin, int aOff0, int aOff1,
                                               int bOff, v8f acc) {
  for (int k = kBegin; k < kBegin + KCHUNK; k += 32) {
    v8bf t0 = *(const v8bf*)(Wrow + k + aOff0);
    v8bf t1 = *(const v8bf*)(Wrow + k + aOff1);
    v8bf x0 = *(const v8bf*)(xb + k + bOff);
    v8bf x1 = *(const v8bf*)(xb + k + bOff + 16);
    v16bf a, b;
#pragma unroll
    for (int i = 0; i < 8; ++i) {
      a[i] = t0[i]; a[8 + i] = t1[i];
      b[i] = x0[i]; b[8 + i] = x1[i];
    }
    acc = __builtin_amdgcn_wmma_f32_16x16x32_bf16(false, a, false, b, (short)0, acc, false, false);
  }
  return acc;
}

// g[8192] = Wih*x + Whh*h + bias. Grid: H4/(16*2)=256 WGs, 8 waves each
// (2 tiles per WG, 4 K-split waves per tile) -> 4x more WGPs engaged per
// sequential step than 1-wave-per-tile, same total L2 traffic.
__global__ __launch_bounds__(256) void k_gates(const bf16_t* __restrict__ Wih,
                                               const bf16_t* __restrict__ Whh,
                                               const bf16_t* __restrict__ xb,
                                               const bf16_t* __restrict__ hb,
                                               const float* __restrict__ bias,
                                               float* __restrict__ g) {
  __shared__ float red[2][KSPLIT][16];
  const int lane = threadIdx.x & 31;
  const int wave = threadIdx.x >> 5;
  const int tileInWG = wave >> 2;          // 0..1
  const int kwave = wave & (KSPLIT - 1);   // 0..3
  const int rowBase = (blockIdx.x * 2 + tileInWG) * 16;
  const int kBegin = kwave * KCHUNK;
  const int aOff0 = (lane < 16) ? 0 : 8;   // A half-slot K offsets (ISA layout)
  const int aOff1 = (lane < 16) ? 16 : 24;
  const int am = lane & 15;
  const int bOff = (lane < 16) ? 0 : 8;    // B: lanes0-15 K0..7,16..23, lanes16-31 K8..15,24..31
  const bf16_t* wihRow = Wih + (size_t)(rowBase + am) * H;
  const bf16_t* whhRow = Whh + (size_t)(rowBase + am) * H;
  v8f acc = {};
  acc = wmma_gemv_chunk(wihRow, xb, kBegin, aOff0, aOff1, bOff, acc);
  acc = wmma_gemv_chunk(whhRow, hb, kBegin, aOff0, aOff1, bOff, acc);
  if (lane < 8)                      red[tileInWG][kwave][lane] = acc[lane];
  else if (lane >= 16 && lane < 24)  red[tileInWG][kwave][8 + lane - 16] = acc[lane - 16];
  __syncthreads();
  if (threadIdx.x < 32) {
    int tw = threadIdx.x >> 4, r = threadIdx.x & 15;
    float s = red[tw][0][r] + red[tw][1][r] + red[tw][2][r] + red[tw][3][r];
    int row = (blockIdx.x * 2 + tw) * 16 + r;
    g[row] = s + bias[row];
  }
}

// wh = attn1_W * h ; att2h = attn2_W * h (2048x2048 GEMVs, same scheme)
__global__ __launch_bounds__(256) void k_attn(const bf16_t* __restrict__ A1,
                                              const bf16_t* __restrict__ A2,
                                              const bf16_t* __restrict__ hb,
                                              float* __restrict__ wh_slot,
                                              float* __restrict__ att2h) {
  __shared__ float red1[2][KSPLIT][16];
  __shared__ float red2[2][KSPLIT][16];
  const int lane = threadIdx.x & 31;
  const int wave = threadIdx.x >> 5;
  const int tileInWG = wave >> 2;
  const int kwave = wave & (KSPLIT - 1);
  const int rowBase = (blockIdx.x * 2 + tileInWG) * 16;
  const int kBegin = kwave * KCHUNK;
  const int aOff0 = (lane < 16) ? 0 : 8;
  const int aOff1 = (lane < 16) ? 16 : 24;
  const int am = lane & 15;
  const int bOff = (lane < 16) ? 0 : 8;
  const bf16_t* r1 = A1 + (size_t)(rowBase + am) * H;
  const bf16_t* r2 = A2 + (size_t)(rowBase + am) * H;
  v8f acc1 = {}, acc2 = {};
  acc1 = wmma_gemv_chunk(r1, hb, kBegin, aOff0, aOff1, bOff, acc1);
  acc2 = wmma_gemv_chunk(r2, hb, kBegin, aOff0, aOff1, bOff, acc2);
  if (lane < 8) {
    red1[tileInWG][kwave][lane] = acc1[lane];
    red2[tileInWG][kwave][lane] = acc2[lane];
  } else if (lane >= 16 && lane < 24) {
    red1[tileInWG][kwave][8 + lane - 16] = acc1[lane - 16];
    red2[tileInWG][kwave][8 + lane - 16] = acc2[lane - 16];
  }
  __syncthreads();
  if (threadIdx.x < 32) {
    int tw = threadIdx.x >> 4, r = threadIdx.x & 15;
    float s1 = red1[tw][0][r] + red1[tw][1][r] + red1[tw][2][r] + red1[tw][3][r];
    float s2 = red2[tw][0][r] + red2[tw][1][r] + red2[tw][2][r] + red2[tw][3][r];
    int row = (blockIdx.x * 2 + tw) * 16 + r;
    wh_slot[row] = s1; att2h[row] = s2;
  }
}

// LSTM pointwise: i,f,g,o gates -> h2, c2 (and optional append into all_h)
__global__ __launch_bounds__(256) void k_pointwise(const float* __restrict__ g,
                                                   float* __restrict__ h, float* __restrict__ c,
                                                   bf16_t* __restrict__ hb,
                                                   float* __restrict__ appendRow) {
  int i = blockIdx.x * blockDim.x + threadIdx.x;
  if (i >= H) return;
  float gi = g[i], gf = g[H + i], gg = g[2 * H + i], go = g[3 * H + i];
  float si = 1.f / (1.f + __expf(-gi));
  float sf = 1.f / (1.f + __expf(-gf));
  float so = 1.f / (1.f + __expf(-go));
  float c2 = sf * c[i] + si * tanhf(gg);
  float h2 = so * tanhf(c2);
  c[i] = c2; h[i] = h2; hb[i] = (bf16_t)h2;
  if (appendRow) appendRow[i] = h2;
}

// score over L previous states, uniform-categorical sample (reference's
// softmax(dim=0) degenerates to uniform), then embed = all_h[skip]
__global__ __launch_bounds__(256) void k_score_sample(const float* __restrict__ all_wh,
                                                      const float* __restrict__ att2h,
                                                      const float* __restrict__ attnv,
                                                      const float* __restrict__ all_h,
                                                      int L, unsigned* __restrict__ rng,
                                                      float* __restrict__ seq_out,
                                                      float* __restrict__ embed,
                                                      bf16_t* __restrict__ embed_bf,
                                                      float* __restrict__ score_out) {
  __shared__ float red[256];
  __shared__ int s_skip;
  for (int l = 0; l < L; ++l) {
    float p = 0.f;
    const float* wh = all_wh + (size_t)l * H;
    for (int i = threadIdx.x; i < H; i += 256) p += tanhf(wh[i] + att2h[i]) * attnv[i];
    red[threadIdx.x] = p;
    __syncthreads();
    for (int s2 = 128; s2 > 0; s2 >>= 1) {
      if (threadIdx.x < (unsigned)s2) red[threadIdx.x] += red[threadIdx.x + s2];
      __syncthreads();
    }
    if (threadIdx.x == 0) score_out[l] = red[0];
    __syncthreads();
  }
  if (threadIdx.x == 0) {
    unsigned s = *rng; s = s * 1664525u + 1013904223u; *rng = s;
    int skip = (int)((s >> 16) % (unsigned)L);
    s_skip = skip;
    seq_out[0] = (float)skip;
  }
  __syncthreads();
  const float* src = all_h + (size_t)s_skip * H;
  for (int i = threadIdx.x; i < H; i += 256) {
    float v = src[i];
    embed[i] = v; embed_bf[i] = (bf16_t)v;
  }
}

// op step: logit = lin_W*h + lin_b (6x2048 matvec), write logits + sampled op
__global__ __launch_bounds__(256) void k_oplogits(const float* __restrict__ h,
                                                  const float* __restrict__ linW,
                                                  const float* __restrict__ linb,
                                                  float* __restrict__ logit_out,
                                                  float* __restrict__ seq_out,
                                                  unsigned* __restrict__ rng) {
  __shared__ float red[256];
  for (int cc = 0; cc < NCELLS; ++cc) {
    float p = 0.f;
    const float* w = linW + (size_t)cc * H;
    for (int i = threadIdx.x; i < H; i += 256) p += h[i] * w[i];
    red[threadIdx.x] = p;
    __syncthreads();
    for (int s2 = 128; s2 > 0; s2 >>= 1) {
      if (threadIdx.x < (unsigned)s2) red[threadIdx.x] += red[threadIdx.x + s2];
      __syncthreads();
    }
    if (threadIdx.x == 0) logit_out[cc] = red[0] + linb[cc];
    __syncthreads();
  }
  if (threadIdx.x == 0) {
    unsigned s = *rng; s = s * 1664525u + 1013904223u; *rng = s;
    seq_out[0] = (float)((s >> 16) % (unsigned)NCELLS);
  }
}

extern "C" void kernel_launch(void* const* d_in, const int* in_sizes, int n_in,
                              void* d_out, int out_size, void* d_ws, size_t ws_size,
                              hipStream_t stream) {
  (void)in_sizes; (void)n_in; (void)out_size; (void)ws_size;
  const float* emb0 = (const float*)d_in[0];
  const float* Wih  = (const float*)d_in[1];
  const float* Whh  = (const float*)d_in[2];
  const float* bih  = (const float*)d_in[3];
  const float* bhh  = (const float*)d_in[4];
  const float* linW = (const float*)d_in[5];
  const float* linb = (const float*)d_in[6];
  const float* a1   = (const float*)d_in[7];
  const float* a2   = (const float*)d_in[8];
  const float* av   = (const float*)d_in[9];
  float* out = (float*)d_out;

  // workspace carve-out (~85 MB: bf16 weights dominate; fits MI455X 192MB L2)
  char* p = (char*)d_ws;
  auto alloc = [&](size_t bytes) -> char* {
    char* r = p; p += (bytes + 255) & ~(size_t)255; return r;
  };
  bf16_t* wih_b = (bf16_t*)alloc((size_t)H4 * H * 2);
  bf16_t* whh_b = (bf16_t*)alloc((size_t)H4 * H * 2);
  bf16_t* a1_b  = (bf16_t*)alloc((size_t)H * H * 2);
  bf16_t* a2_b  = (bf16_t*)alloc((size_t)H * H * 2);
  float*  bias  = (float*)alloc(H4 * 4);
  float*  g     = (float*)alloc(H4 * 4);
  float*  hbuf  = (float*)alloc(H * 4);
  float*  cbuf  = (float*)alloc(H * 4);
  float*  embed = (float*)alloc(H * 4);
  bf16_t* emb_b = (bf16_t*)alloc(H * 2);
  bf16_t* h_b   = (bf16_t*)alloc(H * 2);
  float*  att2h = (float*)alloc(H * 4);
  float*  all_h = (float*)alloc((size_t)48 * H * 4);
  float*  all_wh= (float*)alloc((size_t)48 * H * 4);
  float*  scores= (float*)alloc(32 * 4);
  unsigned* rng = (unsigned*)alloc(256);

  auto cvt = [&](const float* s, bf16_t* d, int n) {
    int blk = (n + 256 * 8 - 1) / (256 * 8);
    hipLaunchKernelGGL(k_convert, dim3(blk), dim3(256), 0, stream, s, d, n);
  };
  cvt(Wih, wih_b, H4 * H);
  cvt(Whh, whh_b, H4 * H);
  cvt(a1, a1_b, H * H);
  cvt(a2, a2_b, H * H);
  hipLaunchKernelGGL(k_init, dim3(32), dim3(256), 0, stream,
                     emb0, bih, bhh, hbuf, cbuf, embed, emb_b, h_b, bias, rng);

  const dim3 gatesGrid(H4 / (16 * 2));   // 256 WGs: 2 tiles/WG, 4 K-waves/tile
  const dim3 attnGrid(H / (16 * 2));     // 64 WGs
  const dim3 pwGrid(H / 256);

  int h_count = 0;
  // warmup layers 0,1: lstm step + append h and attn1*h
  for (int s = 0; s < 2; ++s) {
    hipLaunchKernelGGL(k_gates, gatesGrid, dim3(256), 0, stream,
                       wih_b, whh_b, emb_b, h_b, bias, g);
    hipLaunchKernelGGL(k_pointwise, pwGrid, dim3(256), 0, stream,
                       g, hbuf, cbuf, h_b, all_h + (size_t)h_count * H);
    hipLaunchKernelGGL(k_attn, attnGrid, dim3(256), 0, stream,
                       a1_b, a2_b, h_b, all_wh + (size_t)h_count * H, att2h);
    ++h_count;
  }
  for (int layer = 2; layer < NUM_LAYERS; ++layer) {
    int li = layer - 2;
    for (int s = 0; s < 2; ++s) {           // skip-connection sampling steps
      hipLaunchKernelGGL(k_gates, gatesGrid, dim3(256), 0, stream,
                         wih_b, whh_b, emb_b, h_b, bias, g);
      hipLaunchKernelGGL(k_pointwise, pwGrid, dim3(256), 0, stream,
                         g, hbuf, cbuf, h_b, all_h + (size_t)h_count * H);
      hipLaunchKernelGGL(k_attn, attnGrid, dim3(256), 0, stream,
                         a1_b, a2_b, h_b, all_wh + (size_t)h_count * H, att2h);
      hipLaunchKernelGGL(k_score_sample, dim3(1), dim3(256), 0, stream,
                         all_wh, att2h, av, all_h, layer, rng,
                         out + (li * 4 + s), embed, emb_b, scores);
      ++h_count;
    }
    for (int t = 0; t < 2; ++t) {           // op sampling steps (embed unchanged)
      hipLaunchKernelGGL(k_gates, gatesGrid, dim3(256), 0, stream,
                         wih_b, whh_b, emb_b, h_b, bias, g);
      hipLaunchKernelGGL(k_pointwise, pwGrid, dim3(256), 0, stream,
                         g, hbuf, cbuf, h_b, (float*)nullptr);
      int opRow = li * 2 + t;
      hipLaunchKernelGGL(k_oplogits, dim3(1), dim3(256), 0, stream,
                         hbuf, linW, linb,
                         out + 88 + opRow * NCELLS,
                         out + (li * 4 + 2 + t), rng);
    }
  }
}
